// GRU_26336739459279
// MI455X (gfx1250) — matrix operand order repaired and checked by reading, not compiled, not run on hardware
//
#include <hip/hip_runtime.h>
#include <hip/hip_bf16.h>
#include <math.h>

// ---------------------------------------------------------------------------
// Problem constants (fixed by the reference)
// ---------------------------------------------------------------------------
#define B_    4
#define S_    2048
#define DIN_  2048
#define ST_   2048
#define H_    32
#define HD_   64
#define TS_   6144        // 3 * STATE
#define BS_   8192        // B * S
#define EPS_  1e-6f

typedef __bf16 bf16_t;
typedef __attribute__((ext_vector_type(16))) __bf16 v16bf;
typedef __attribute__((ext_vector_type(8)))  __bf16 v8bf;
typedef __attribute__((ext_vector_type(8)))  float   v8f;
typedef __attribute__((ext_vector_type(4)))  float   v4f;
typedef __attribute__((ext_vector_type(4)))  int     v4i;

// generic -> hard address-space casts (numeric identity: LDS generic addr low
// 32 bits are the LDS byte offset; global generic == AS1). The async-LDS
// builtin wants v4i* operands (probe: param 1 is "__device__ v4i *").
#define TO_GLOBAL(p) ((__attribute__((address_space(1))) v4i*)(unsigned long long)(p))
#define TO_LDS(p)    ((__attribute__((address_space(3))) v4i*)(unsigned int)(unsigned long long)(p))

#if defined(__gfx1250__) && __has_builtin(__builtin_amdgcn_global_load_async_to_lds_b128)
#define HAVE_ASYNC_LDS 1
#else
#define HAVE_ASYNC_LDS 0
#endif

// ---------------------------------------------------------------------------
// f32 -> bf16 conversion, 8 elements / thread, b128 loads+stores
// ---------------------------------------------------------------------------
__global__ __launch_bounds__(256) void k_f32_to_bf16(const float* __restrict__ x,
                                                     bf16_t* __restrict__ y,
                                                     long n8) {
  long i = (long)blockIdx.x * blockDim.x + threadIdx.x;
  if (i >= n8) return;
  const v4f* xp = (const v4f*)x + i * 2;
  v4f a = xp[0];
  v4f b = xp[1];
  v8bf o;
  o[0] = (bf16_t)a[0]; o[1] = (bf16_t)a[1]; o[2] = (bf16_t)a[2]; o[3] = (bf16_t)a[3];
  o[4] = (bf16_t)b[0]; o[5] = (bf16_t)b[1]; o[6] = (bf16_t)b[2]; o[7] = (bf16_t)b[3];
  *((v8bf*)y + i) = o;
}

// ---------------------------------------------------------------------------
// p-norm (p=2, no affine) over each flattened [HD*HD]=4096 row of [3H,...]
// ---------------------------------------------------------------------------
__global__ __launch_bounds__(256) void k_pnorm_rows(const float* __restrict__ w,
                                                    float* __restrict__ out) {
  __shared__ float red[8];
  const int row = blockIdx.x;                  // 0..95
  const float* r = w + (size_t)row * 4096;
  float vals[16];
  float ss = 0.f;
#pragma unroll
  for (int i = 0; i < 16; ++i) {
    float v = r[threadIdx.x + i * 256];
    vals[i] = v;
    ss += v * v;
  }
#pragma unroll
  for (int off = 16; off > 0; off >>= 1) ss += __shfl_down(ss, off, 32);
  if ((threadIdx.x & 31) == 0) red[threadIdx.x >> 5] = ss;
  __syncthreads();
  float tot = 0.f;
#pragma unroll
  for (int i = 0; i < 8; ++i) tot += red[i];
  const float scale = rsqrtf(tot * (1.0f / 4096.0f) + EPS_);
#pragma unroll
  for (int i = 0; i < 16; ++i)
    out[(size_t)row * 4096 + threadIdx.x + i * 256] = vals[i] * scale;
}

// ---------------------------------------------------------------------------
// WMMA bf16 GEMM:  C[M,N] = A[M,K] * W[N,K]^T   (both row-major, K-major dot)
// Block: 256 threads = 8 waves. Wave tile 32M x 64N (2 A-frags x 4 B-frags ->
// 8 WMMA per 12 b128 loads). Block tile 256M x 64N.
// Software-pipelined (ping-pong fragment sets over K, step 64) so fragment
// loads for chunk k+1 are in flight while WMMAs consume chunk k.
// Fragment layouts per CDNA5 ISA 7.12.2 (wave32).
// ---------------------------------------------------------------------------
union FragU { v16bf v; v8bf g[2]; };

__device__ __forceinline__ void gemm_load_set(FragU a[2], FragU b[4],
                                              const bf16_t* __restrict__ arow0,
                                              const bf16_t* __restrict__ arow1,
                                              const bf16_t* __restrict__ wrow,
                                              size_t wstride16, int k) {
  a[0].g[0] = *(const v8bf*)(arow0 + k);
  a[0].g[1] = *(const v8bf*)(arow0 + k + 16);
  a[1].g[0] = *(const v8bf*)(arow1 + k);
  a[1].g[1] = *(const v8bf*)(arow1 + k + 16);
#pragma unroll
  for (int ni = 0; ni < 4; ++ni) {
    const bf16_t* wp = wrow + ni * wstride16 + k;
    b[ni].g[0] = *(const v8bf*)(wp);
    b[ni].g[1] = *(const v8bf*)(wp + 16);
  }
}

__device__ __forceinline__ void gemm_mma_set(v8f acc[8], const FragU a[2], const FragU b[4]) {
#pragma unroll
  for (int mi = 0; mi < 2; ++mi)
#pragma unroll
    for (int ni = 0; ni < 4; ++ni)
      acc[mi * 4 + ni] = __builtin_amdgcn_wmma_f32_16x16x32_bf16(
          false, a[mi].v, false, b[ni].v, (short)0, acc[mi * 4 + ni], false, false);
}

template <typename OutT>
__global__ __launch_bounds__(256) void k_wmma_gemm_nt(const bf16_t* __restrict__ A,
                                                      const bf16_t* __restrict__ W,
                                                      OutT* __restrict__ C,
                                                      int M, int N, int K) {
  const int wave = threadIdx.x >> 5;
  const int lane = threadIdx.x & 31;
  const int half = lane >> 4;     // lane group 0/1
  const int l16  = lane & 15;

  const int tileM = (blockIdx.x * 8 + wave) * 32;
  const int tileN = blockIdx.y * 64;
  if (tileM >= M) return;   // wave-uniform: EXEC stays all-ones for WMMA

  v8f acc[8];
#pragma unroll
  for (int i = 0; i < 8; ++i) acc[i] = (v8f){};

  // A fragment 16x32: row = l16; elems 0..7  <- K = half*8 + 0..7
  //                              elems 8..15 <- K = 16 + half*8 + 0..7
  const bf16_t* arow0 = A + (size_t)(tileM + l16) * K + half * 8;
  const bf16_t* arow1 = arow0 + (size_t)16 * K;
  // B fragment 32x16: col = l16; elems 0..7 <- K = half*8 + 0..7, elems 8..15 <- K = 16 + half*8 + 0..7
  const bf16_t* wrow = W + (size_t)(tileN + l16) * K + half * 8;
  const size_t wstride16 = (size_t)16 * K;   // 16 N-rows forward

  FragU aP[2], bP[4], aQ[2], bQ[4];

  gemm_load_set(aP, bP, arow0, arow1, wrow, wstride16, 0);
  // K is a multiple of 64 for all uses here (K = 2048)
  for (int k = 0; k < K - 64; k += 64) {
    gemm_load_set(aQ, bQ, arow0, arow1, wrow, wstride16, k + 32);
    gemm_mma_set(acc, aP, bP);
    gemm_load_set(aP, bP, arow0, arow1, wrow, wstride16, k + 64);
    gemm_mma_set(acc, aQ, bQ);
  }
  gemm_load_set(aQ, bQ, arow0, arow1, wrow, wstride16, K - 32);
  gemm_mma_set(acc, aP, bP);
  gemm_mma_set(acc, aQ, bQ);

  // D layout: lane group half -> rows M = half*8 + v (v = VGPR index), col = l16
#pragma unroll
  for (int mi = 0; mi < 2; ++mi) {
#pragma unroll
    for (int v = 0; v < 8; ++v) {
      const int mrow = tileM + mi * 16 + half * 8 + v;
      OutT* crow = C + (size_t)mrow * N + tileN + l16;
      crow[0]  = (OutT)acc[mi * 4 + 0][v];
      crow[16] = (OutT)acc[mi * 4 + 1][v];
      crow[32] = (OutT)acc[mi * 4 + 2][v];
      crow[48] = (OutT)acc[mi * 4 + 3][v];
    }
  }
}

// ---------------------------------------------------------------------------
// Causal depthwise conv1d (K=4) + SiLU + RMSNorm (per gate of 2048 channels)
// Block = one (b,s,gate); 256 threads x 8 channels.
// ---------------------------------------------------------------------------
__global__ __launch_bounds__(256) void k_conv_silu_rmsnorm(
    const bf16_t* __restrict__ p,       // [B*S, 6144] bf16
    const float* __restrict__ conv_w,   // [6144, 4]
    const float* __restrict__ nw0,
    const float* __restrict__ nw1,
    const float* __restrict__ nw2,
    float* __restrict__ gates) {        // [B*S, 6144] f32
  __shared__ float red[8];
  const int bs = blockIdx.x;            // b*S + s
  const int g  = blockIdx.y;            // gate 0..2
  const int s  = bs & (S_ - 1);
  const int cg = threadIdx.x * 8;       // channel within gate
  const int c  = g * ST_ + cg;          // absolute channel in [0,6144)

  float acc[8] = {0.f, 0.f, 0.f, 0.f, 0.f, 0.f, 0.f, 0.f};
#pragma unroll
  for (int k = 0; k < 4; ++k) {
    const int st = s - 3 + k;
    if (st >= 0) {
      v8bf pv = *(const v8bf*)(p + (size_t)(bs - 3 + k) * TS_ + c);
#pragma unroll
      for (int j = 0; j < 8; ++j)
        acc[j] = fmaf((float)pv[j], conv_w[(size_t)(c + j) * 4 + k], acc[j]);
    }
  }

  float v[8];
  float ss = 0.f;
#pragma unroll
  for (int j = 0; j < 8; ++j) {
    const float x = acc[j];
    const float sg = 1.f / (1.f + __expf(-x));
    v[j] = x * sg;                       // SiLU
    ss += v[j] * v[j];
  }
#pragma unroll
  for (int off = 16; off > 0; off >>= 1) ss += __shfl_down(ss, off, 32);
  if ((threadIdx.x & 31) == 0) red[threadIdx.x >> 5] = ss;
  __syncthreads();
  float tot = 0.f;
#pragma unroll
  for (int i = 0; i < 8; ++i) tot += red[i];
  const float scale = rsqrtf(tot * (1.0f / (float)ST_) + EPS_);

  const float* nw = (g == 0) ? nw0 : (g == 1) ? nw1 : nw2;
  float* orow = gates + (size_t)bs * TS_ + c;
#pragma unroll
  for (int j = 0; j < 8; ++j)
    orow[j] = v[j] * scale * nw[cg + j];
}

// ---------------------------------------------------------------------------
// Sequential nonlinear recurrence. One 64-thread block per (b, head).
// W / Wf / Wr (48 KB) pulled into LDS with global_load_async_to_lds_b128
// (ASYNCcnt path) when available. State broadcast via LDS b128 reads; 4-way
// split accumulators shorten the serial FMA chain. Next-step gates prefetched.
// ---------------------------------------------------------------------------
__global__ __launch_bounds__(64) void k_rnn_scan(const float* __restrict__ gates, // [B*S,6144]
                                                 const float* __restrict__ sw,    // [96,64,64]
                                                 float* __restrict__ hs) {        // [B*S,2048]
  const int b    = blockIdx.x >> 5;     // 0..3
  const int head = blockIdx.x & 31;     // 0..31
  const int e    = threadIdx.x;         // 0..63

  // [0,4096) = W, [4096,8192) = Wf, [8192,12288) = Wr
  __shared__ __align__(16) float sWall[3 * HD_ * HD_];
  __shared__ __align__(16) float sh[HD_];
  __shared__ __align__(16) float rh[HD_];

#if HAVE_ASYNC_LDS
#pragma unroll
  for (int t = 0; t < 3; ++t) {
    const float* src = sw + (size_t)(t * 32 + head) * 4096;   // row order: W, Wf, Wr
    float* dst = &sWall[t * 4096];
    for (int i = threadIdx.x; i < 1024; i += 64) {            // 1024 x 16B = 16KB
      __builtin_amdgcn_global_load_async_to_lds_b128(
          TO_GLOBAL(src + i * 4), TO_LDS(dst + i * 4), 0, 0);
    }
  }
#if __has_builtin(__builtin_amdgcn_s_wait_asynccnt)
  __builtin_amdgcn_s_wait_asynccnt(0);
#else
  asm volatile("s_wait_asynccnt 0x0" ::: "memory");
#endif
#else
  for (int i = threadIdx.x; i < HD_ * HD_; i += 64) {
    sWall[i]        = sw[(size_t)head * 4096 + i];
    sWall[4096 + i] = sw[(size_t)(32 + head) * 4096 + i];
    sWall[8192 + i] = sw[(size_t)(64 + head) * 4096 + i];
  }
#endif
  sh[e] = 0.f;
  __syncthreads();

  float h = 0.f;
  const float* gb = gates + (size_t)b * S_ * TS_ + head * HD_ + e;
  float it = gb[0];
  float ft = gb[ST_];
  float rt = gb[2 * ST_];

  for (int s = 0; s < S_; ++s) {
    // prefetch next step's gates (hidden behind the matvec loops)
    float itn = 0.f, ftn = 0.f, rtn = 0.f;
    if (s + 1 < S_) {
      const float* gn = gb + (size_t)(s + 1) * TS_;
      itn = gn[0];
      ftn = gn[ST_];
      rtn = gn[2 * ST_];
    }

    float accf, accr;
    {
      float af[4] = {0.f, 0.f, 0.f, 0.f};
      float ar[4] = {0.f, 0.f, 0.f, 0.f};
#pragma unroll
      for (int d = 0; d < HD_; d += 4) {
        v4f hv = *(const v4f*)(sh + d);
#pragma unroll
        for (int q = 0; q < 4; ++q) {
          af[q] = fmaf(hv[q], sWall[4096 + (d + q) * HD_ + e], af[q]);
          ar[q] = fmaf(hv[q], sWall[8192 + (d + q) * HD_ + e], ar[q]);
        }
      }
      accf = (af[0] + af[1]) + (af[2] + af[3]);
      accr = (ar[0] + ar[1]) + (ar[2] + ar[3]);
    }
    const float f = 1.f / (1.f + __expf(-(ft + accf)));
    const float r = 1.f / (1.f + __expf(-(rt + accr)));

    __syncthreads();            // sh reads & previous rh reads complete
    rh[e] = r * h;
    __syncthreads();            // rh visible

    float accn;
    {
      float an[4] = {0.f, 0.f, 0.f, 0.f};
#pragma unroll
      for (int d = 0; d < HD_; d += 4) {
        v4f rv = *(const v4f*)(rh + d);
#pragma unroll
        for (int q = 0; q < 4; ++q)
          an[q] = fmaf(rv[q], sWall[(d + q) * HD_ + e], an[q]);
      }
      accn = (an[0] + an[1]) + (an[2] + an[3]);
    }

    const float n  = tanhf(it + accn);
    const float hn = f * h + (1.f - f) * n;

    hs[((size_t)b * S_ + s) * ST_ + head * HD_ + e] = hn;
    h = hn;
    sh[e] = hn;
    __syncthreads();            // sh(h_s) visible before next iteration reads

    it = itn; ft = ftn; rt = rtn;
  }
}

// ---------------------------------------------------------------------------
// Output RMSNorm + f32 -> bf16 (feeds final WMMA GEMM)
// ---------------------------------------------------------------------------
__global__ __launch_bounds__(256) void k_out_rmsnorm_bf16(const float* __restrict__ hsv,
                                                          const float* __restrict__ w,
                                                          bf16_t* __restrict__ y) {
  __shared__ float red[8];
  const int bs = blockIdx.x;
  const float* r = hsv + (size_t)bs * ST_ + threadIdx.x * 8;
  v4f a = *(const v4f*)r;
  v4f b = *(const v4f*)(r + 4);
  float vals[8] = {a[0], a[1], a[2], a[3], b[0], b[1], b[2], b[3]};
  float ss = 0.f;
#pragma unroll
  for (int j = 0; j < 8; ++j) ss += vals[j] * vals[j];
#pragma unroll
  for (int off = 16; off > 0; off >>= 1) ss += __shfl_down(ss, off, 32);
  if ((threadIdx.x & 31) == 0) red[threadIdx.x >> 5] = ss;
  __syncthreads();
  float tot = 0.f;
#pragma unroll
  for (int i = 0; i < 8; ++i) tot += red[i];
  const float scale = rsqrtf(tot * (1.0f / (float)ST_) + EPS_);

  const float* wp = w + threadIdx.x * 8;
  v8bf o;
#pragma unroll
  for (int j = 0; j < 8; ++j) o[j] = (bf16_t)(vals[j] * scale * wp[j]);
  *(v8bf*)(y + (size_t)bs * ST_ + threadIdx.x * 8) = o;
}

// ---------------------------------------------------------------------------
// Launch
// ---------------------------------------------------------------------------
extern "C" void kernel_launch(void* const* d_in, const int* in_sizes, int n_in,
                              void* d_out, int out_size, void* d_ws, size_t ws_size,
                              hipStream_t stream) {
  (void)in_sizes; (void)n_in; (void)out_size; (void)ws_size;

  const float* input   = (const float*)d_in[0];  // [B,S,DIN]
  const float* w_in    = (const float*)d_in[1];  // [3*ST, DIN]
  const float* conv_w  = (const float*)d_in[2];  // [3*ST,1,4]
  const float* stw     = (const float*)d_in[3];  // [3H,HD,HD]
  const float* in_nw   = (const float*)d_in[4];
  const float* fo_nw   = (const float*)d_in[5];
  const float* re_nw   = (const float*)d_in[6];
  const float* out_nw  = (const float*)d_in[7];
  const float* w_out   = (const float*)d_in[8];  // [ST, ST]
  float* out = (float*)d_out;

  char* ws = (char*)d_ws;
  size_t off = 0;
  auto alloc = [&](size_t bytes) -> void* {
    void* p = ws + off;
    off = (off + bytes + 255) & ~(size_t)255;
    return p;
  };
  bf16_t* a_bf    = (bf16_t*)alloc((size_t)BS_ * DIN_ * 2);   // input bf16
  bf16_t* win_bf  = (bf16_t*)alloc((size_t)TS_ * DIN_ * 2);   // w_in bf16
  bf16_t* wout_bf = (bf16_t*)alloc((size_t)ST_ * ST_ * 2);    // w_out bf16
  bf16_t* p_bf    = (bf16_t*)alloc((size_t)BS_ * TS_ * 2);    // projection out
  float*  gates   = (float*)alloc((size_t)BS_ * TS_ * 4);     // normalized gates
  float*  sw      = (float*)alloc((size_t)96 * 4096 * 4);     // p-normed state W
  float*  hsbuf   = (float*)alloc((size_t)BS_ * ST_ * 4);     // scan outputs
  bf16_t* y_bf    = (bf16_t*)alloc((size_t)BS_ * ST_ * 2);    // normed y bf16

  // 1) precision conversions
  {
    long n8;
    n8 = (long)BS_ * DIN_ / 8;
    k_f32_to_bf16<<<(unsigned)((n8 + 255) / 256), 256, 0, stream>>>(input, a_bf, n8);
    n8 = (long)TS_ * DIN_ / 8;
    k_f32_to_bf16<<<(unsigned)((n8 + 255) / 256), 256, 0, stream>>>(w_in, win_bf, n8);
    n8 = (long)ST_ * ST_ / 8;
    k_f32_to_bf16<<<(unsigned)((n8 + 255) / 256), 256, 0, stream>>>(w_out, wout_bf, n8);
  }

  // 2) p-norm of state transition matrices
  k_pnorm_rows<<<96, 256, 0, stream>>>(stw, sw);

  // 3) input projection GEMM (bf16 WMMA, f32 accumulate, bf16 out)
  k_wmma_gemm_nt<bf16_t><<<dim3(BS_ / 256, TS_ / 64), 256, 0, stream>>>(
      a_bf, win_bf, p_bf, BS_, TS_, DIN_);

  // 4) conv + SiLU + per-gate RMSNorm
  k_conv_silu_rmsnorm<<<dim3(BS_, 3), 256, 0, stream>>>(
      p_bf, conv_w, in_nw, fo_nw, re_nw, gates);

  // 5) sequential recurrence
  k_rnn_scan<<<B_ * H_, 64, 0, stream>>>(gates, sw, hsbuf);

  // 6) output RMSNorm -> bf16
  k_out_rmsnorm_bf16<<<BS_, 256, 0, stream>>>(hsbuf, out_nw, y_bf);

  // 7) output projection GEMM (bf16 WMMA, f32 out)
  k_wmma_gemm_nt<float><<<dim3(BS_ / 256, ST_ / 64), 256, 0, stream>>>(
      y_bf, wout_bf, out, BS_, ST_, DIN_);
}
